// NSSTDecomposition_55886114456111
// MI455X (gfx1250) — matrix operand order repaired and checked by reading, not compiled, not run on hardware
//
#include <hip/hip_runtime.h>

typedef __attribute__((ext_vector_type(16))) _Float16 v16h;
typedef __attribute__((ext_vector_type(8)))  float    v8f;

__device__ __forceinline__ int iclamp(int v, int lo, int hi) {
    return v < lo ? lo : (v > hi ? hi : v);
}

// ---------------------------------------------------------------------------
// Implicit-GEMM grouped convolution via v_wmma_f32_16x16x32_f16.
//   M = direction (padded to 16), N = 16 contiguous output pixels per wave,
//   K = RPC kernel rows packed into 32 taps:
//     RPC=2 (ksize<=16): K 0..15 = row 2c, K 16..31 = row 2c+1  (no K waste)
//     RPC=1 (ksize<=32): K = one kernel row zero-padded to 32
// One workgroup (8 waves) computes an 8-row x 128-col output tile for one
// (batch, color). Filters and the input tile are staged in LDS as f16.
// ---------------------------------------------------------------------------
template <int D, int KS, int RPC>
__global__ __launch_bounds__(256) void conv_wmma_kernel(
    const float* __restrict__ img,   // [B,3,512,512]
    const float* __restrict__ filt,  // [3*D,1,KS,KS] (first D are the dirs)
    float* __restrict__ out,         // [B,87,512,512]
    int ch_base)                     // sub-channel base within the 29-group
{
    constexpr int H = 512, W = 512;
    constexpr int P = KS / 2;
    constexpr int CHUNKS = (KS + RPC - 1) / RPC;   // K-chunks of 32 taps
    constexpr int TILE_W = 128;                    // 8 waves * 16 pixels
    constexpr int TILE_H = 8;
    constexpr int IN_W   = TILE_W + 32;            // sliding window slack
    constexpr int IN_H   = TILE_H + RPC * CHUNKS - 1;  // covers padded rows

    __shared__ _Float16 sFilt[16 * CHUNKS * 32];   // [dir16][chunk][32 taps]
    __shared__ _Float16 sIn[IN_H * IN_W];          // input tile, zero padded

    const int tid  = threadIdx.x;
    const int lane = tid & 31;
    const int wave = tid >> 5;

    const int x0 = blockIdx.x * TILE_W;
    const int y0 = blockIdx.y * TILE_H;
    const int bc = blockIdx.z;               // b*3 + c
    const int b  = bc / 3;
    const int c  = bc % 3;

    // ---- stage filters (f32 -> f16) into packed-chunk layout ----
    for (int i = tid; i < 16 * CHUNKS * 32; i += 256) {
        const int slot  = i & 31;
        const int rest  = i >> 5;
        const int chunk = rest % CHUNKS;
        const int d     = rest / CHUNKS;
        int ky, kx;
        if (RPC == 2) { ky = 2 * chunk + (slot >> 4); kx = slot & 15; }
        else          { ky = chunk;                    kx = slot;      }
        float v = 0.0f;
        if (d < D && ky < KS && kx < KS) v = filt[(d * KS + ky) * KS + kx];
        sFilt[i] = (_Float16)v;
    }

    // ---- stage input tile with zero border padding ----
    const float* imgc = img + (size_t)bc * H * W;
    for (int i = tid; i < IN_H * IN_W; i += 256) {
        const int cx = i % IN_W;
        const int ry = i / IN_W;
        const int gx = x0 - P + cx;
        const int gy = y0 - P + ry;
        float v = 0.0f;
        if (gx >= 0 && gx < W && gy >= 0 && gy < H) v = imgc[gy * W + gx];
        sIn[i] = (_Float16)v;
    }
    __syncthreads();

    const int n     = lane & 15;        // output pixel within wave's strip
    const int khalf = lane >> 4;        // 0: K 0..15 half, 1: K 16..31 half
    const int xoff  = wave * 16 + n;    // col base inside sIn (-P shifted)

    v8f acc[TILE_H];
#pragma unroll
    for (int r = 0; r < TILE_H; ++r)
#pragma unroll
        for (int e = 0; e < 8; ++e) acc[r][e] = 0.0f;

    // ---- main loop: one A reload + 8 WMMAs per K-chunk ----
    for (int chunk = 0; chunk < CHUNKS; ++chunk) {
        // A matrix, 16x32 f16: lane M = lane&15 holds K {0..15}
        // (khalf=0) or K {16..31} (khalf=1).
        const _Float16* fbase =
            &sFilt[((lane & 15) * CHUNKS + chunk) * 32 + khalf * 16];
        v16h a;
#pragma unroll
        for (int i = 0; i < 8; ++i) a[i] = fbase[i];
#pragma unroll
        for (int i = 0; i < 8; ++i) a[8 + i] = fbase[8 + i];

        // B matrix, 32x16 f16: lane = column n; its 16 K-values are 16
        // consecutive input pixels (Toeplitz window) of one staged row.
        const int rowBase = (RPC == 2) ? (RPC * chunk + khalf) : chunk;
        const int colBase = (RPC == 2) ? xoff : (xoff + khalf * 16);

#pragma unroll
        for (int r = 0; r < TILE_H; ++r) {
            const _Float16* bbase = &sIn[(r + rowBase) * IN_W + colBase];
            v16h bm;
#pragma unroll
            for (int i = 0; i < 16; ++i) bm[i] = bbase[i];

            acc[r] = __builtin_amdgcn_wmma_f32_16x16x32_f16(
                false, a, false, bm, (short)0, acc[r], false, false);
        }
    }

    // ---- write out: C/D layout -> VGPR r8 is M=r8+8*khalf, N=lane&15 ----
    float* outc = out + ((size_t)b * 87 + (size_t)c * 29 + ch_base) * H * W;
    const int gx = x0 + wave * 16 + n;
#pragma unroll
    for (int r8 = 0; r8 < 8; ++r8) {
        const int d = r8 + khalf * 8;
        if (d < D) {
#pragma unroll
            for (int r = 0; r < TILE_H; ++r) {
                outc[(size_t)d * H * W + (size_t)(y0 + r) * W + gx] = acc[r][r8];
            }
        }
    }
}

// ---------------------------------------------------------------------------
// Low band: 4x4 average pool, then bilinear (half-pixel centers, edge clamp).
// ---------------------------------------------------------------------------
__global__ __launch_bounds__(256) void pool4_kernel(
    const float* __restrict__ img, float* __restrict__ pooled)
{
    const int idx = blockIdx.x * 256 + threadIdx.x;     // 12*128*128
    if (idx >= 12 * 128 * 128) return;
    const int x  = idx & 127;
    const int y  = (idx >> 7) & 127;
    const int bc = idx >> 14;
    const float* p = img + (size_t)bc * 512 * 512;
    float s = 0.0f;
#pragma unroll
    for (int dy = 0; dy < 4; ++dy)
#pragma unroll
        for (int dx = 0; dx < 4; ++dx)
            s += p[(y * 4 + dy) * 512 + (x * 4 + dx)];
    pooled[idx] = s * (1.0f / 16.0f);
}

__global__ __launch_bounds__(256) void upsample_kernel(
    const float* __restrict__ pooled, float* __restrict__ out)
{
    const int idx = blockIdx.x * 256 + threadIdx.x;     // 12*512*512
    if (idx >= 12 * 512 * 512) return;
    const int x  = idx & 511;
    const int y  = (idx >> 9) & 511;
    const int bc = idx >> 18;
    const int b  = bc / 3;
    const int c  = bc % 3;
    const float* p = pooled + (size_t)bc * 128 * 128;

    const float fy = (y + 0.5f) * 0.25f - 0.5f;
    const float fx = (x + 0.5f) * 0.25f - 0.5f;
    const int yl = (int)floorf(fy);
    const int xl = (int)floorf(fx);
    const float wy = fy - (float)yl;
    const float wx = fx - (float)xl;
    const int y0c = iclamp(yl, 0, 127), y1c = iclamp(yl + 1, 0, 127);
    const int x0c = iclamp(xl, 0, 127), x1c = iclamp(xl + 1, 0, 127);

    const float v =
        (1.0f - wy) * ((1.0f - wx) * p[y0c * 128 + x0c] + wx * p[y0c * 128 + x1c]) +
        wy          * ((1.0f - wx) * p[y1c * 128 + x0c] + wx * p[y1c * 128 + x1c]);

    out[((size_t)b * 87 + (size_t)c * 29 + 28) * 512 * 512 + (size_t)y * 512 + x] = v;
}

extern "C" void kernel_launch(void* const* d_in, const int* in_sizes, int n_in,
                              void* d_out, int out_size, void* d_ws, size_t ws_size,
                              hipStream_t stream) {
    const float* img = (const float*)d_in[0];
    const float* f0  = (const float*)d_in[1];
    const float* f1  = (const float*)d_in[2];
    const float* f2  = (const float*)d_in[3];
    float* out       = (float*)d_out;
    float* pooled    = (float*)d_ws;     // 12*128*128 floats = 768 KB scratch

    const dim3 blk(256);
    const dim3 grd(512 / 128, 512 / 8, 4 * 3);   // (4, 64, 12)

    // scale0: 4 dirs, 11x11, 2 rows/chunk -> 6 chunks
    conv_wmma_kernel<4, 11, 2><<<grd, blk, 0, stream>>>(img, f0, out, 0);
    // scale1: 8 dirs, 15x15, 2 rows/chunk -> 8 chunks
    conv_wmma_kernel<8, 15, 2><<<grd, blk, 0, stream>>>(img, f1, out, 4);
    // scale2: 16 dirs, 19x19, 1 row/chunk -> 19 chunks
    conv_wmma_kernel<16, 19, 1><<<grd, blk, 0, stream>>>(img, f2, out, 12);

    pool4_kernel<<<(12 * 128 * 128 + 255) / 256, blk, 0, stream>>>(img, pooled);
    upsample_kernel<<<(12 * 512 * 512 + 255) / 256, blk, 0, stream>>>(pooled, out);
}
